// CausalAttention_32882269618920
// MI455X (gfx1250) — matrix operand order repaired and checked by reading, not compiled, not run on hardware
//
#include <hip/hip_runtime.h>

// ---------------------------------------------------------------------------
// CDNA5 (gfx1250) causal attention: f16 WMMA pipeline with f32 accumulation.
//   B=2, N=2048, C=1024, H=16, HD=64
// ---------------------------------------------------------------------------

typedef __attribute__((ext_vector_type(16))) _Float16 v16h;
typedef __attribute__((ext_vector_type(8)))  float    v8f;

union frag16 { v16h h; unsigned int u[8]; };
template <bool V> struct BoolC { static constexpr bool value = V; };

#define BB 2
#define NN 2048
#define CC 1024
#define HH 16
#define HD 64
#define MROWS (BB * NN)   // 4096

// softmax scale folded with log2(e): exp(x*0.125) == exp2(x*0.125*log2e)
#define SM_SCALE 0.18033688f

// ISA 16-bit A-matrix 16x32 layout: VGPR v, lane-half hf -> first K of packed pair
__device__ __forceinline__ int a_koff(int v, int hf) {
    return ((v >> 2) << 4) + hf * 8 + ((v & 3) << 1);
}
// ISA 16-bit B-matrix 32x16 layout: VGPR v, lane-half hf -> first K of packed pair
__device__ __forceinline__ int b_koff(int v, int hf) {
    return ((v >> 2) << 4) + hf * 8 + ((v & 3) << 1);
}

// ---------------------------------------------------------------------------
// f32 -> f16 convert, vectorized x4
// ---------------------------------------------------------------------------
__global__ void cvt_f16_kernel(const float* __restrict__ src,
                               _Float16* __restrict__ dst, int n4) {
    int i = blockIdx.x * 256 + threadIdx.x;
    if (i < n4) {
        float4 f = ((const float4*)src)[i];
        union { _Float16 h[4]; uint2 u; } o;
        o.h[0] = (_Float16)f.x; o.h[1] = (_Float16)f.y;
        o.h[2] = (_Float16)f.z; o.h[3] = (_Float16)f.w;
        ((uint2*)dst)[i] = o.u;
    }
}

// ---------------------------------------------------------------------------
// Tiled GEMM: D = A(f16)[M,K] x B(f16)[K,Nout] (+bias), WMMA f32 accum.
//  MODE 0: scatter f16 output into q/k/v [B,H,N,HD] (Nout = 3C, no bias)
//  MODE 1: +bias, store f32 -> out_f32 AND f16 -> out_f16
//  MODE 2: +bias, store f32 -> out_f32
// Block: 256 threads = 8 waves. Block tile 128x128x32, double-buffered LDS,
// software-pipelined global->regs->LDS. Wave grid 2(M)x4(N): 4x2 WMMA tiles.
// ---------------------------------------------------------------------------
#define BM 128
#define BN 128
#define BK 32

template <int MODE>
__global__ __launch_bounds__(256) void gemm_f16_kernel(
    const _Float16* __restrict__ A, const _Float16* __restrict__ Bm,
    int K, int Nout,
    const float* __restrict__ bias,
    float* __restrict__ out_f32, _Float16* __restrict__ out_f16,
    _Float16* __restrict__ qh, _Float16* __restrict__ kh,
    _Float16* __restrict__ vh)
{
    __shared__ _Float16 As[2][BM][BK];   // natural [m][k]
    __shared__ _Float16 Bt[2][BN][BK];   // transposed [n][k]

    const int tid  = threadIdx.x;
    const int lane = tid & 31;
    const int wave = tid >> 5;
    const int hf   = (lane >> 4) & 1;
    const int ln   = lane & 15;

    const int bm0 = blockIdx.y * BM;
    const int bn0 = blockIdx.x * BN;
    const int wm0 = (wave >> 2) * 64;   // 0 or 64
    const int wn0 = (wave & 3) * 32;    // 0,32,64,96

    // per-thread staging coordinates
    const int ar  = tid >> 1;           // A row 0..127
    const int ac  = (tid & 1) * 16;     // A col 0 or 16
    const int bkr = tid >> 3;           // B k-row 0..31
    const int bcb = (tid & 7) * 16;     // B col 0..112

    const _Float16* aptr = A  + (size_t)(bm0 + ar) * K + ac;
    const _Float16* bptr = Bm + (size_t)bkr * Nout + bn0 + bcb;

    union BU { uint4 u[2]; _Float16 h[16]; };

    v8f vzero = {};
    v8f acc[4][2];
#pragma unroll
    for (int i = 0; i < 4; i++)
#pragma unroll
        for (int j = 0; j < 2; j++) acc[i][j] = vzero;

    // ---- prologue: stage tile 0 into LDS buffer 0 ----
    {
        uint4 a0 = *(const uint4*)aptr;
        uint4 a1 = *(const uint4*)(aptr + 8);
        BU rb;
        rb.u[0] = *(const uint4*)bptr;
        rb.u[1] = *(const uint4*)(bptr + 8);
        *(uint4*)&As[0][ar][ac]     = a0;
        *(uint4*)&As[0][ar][ac + 8] = a1;
#pragma unroll
        for (int e = 0; e < 16; e++) Bt[0][bcb + e][bkr] = rb.h[e];
    }
    __syncthreads();

    const int nk = K / BK;
    int cur = 0;
    for (int t = 0; t < nk; t++) {
        const bool hasNext = (t + 1 < nk);

        // issue next tile's global loads early (latency hidden behind WMMA)
        uint4 na0, na1;
        BU nrb;
        if (hasNext) {
            const _Float16* ap = aptr + (t + 1) * BK;
            na0 = *(const uint4*)ap;
            na1 = *(const uint4*)(ap + 8);
            const _Float16* bp = bptr + (size_t)(t + 1) * BK * Nout;
            nrb.u[0] = *(const uint4*)bp;
            nrb.u[1] = *(const uint4*)(bp + 8);
        }

        // ---- compute current stage ----
        frag16 afr[4], bfr[2];
#pragma unroll
        for (int i = 0; i < 4; i++) {
            int m = wm0 + i * 16 + ln;
#pragma unroll
            for (int v = 0; v < 8; v++)
                afr[i].u[v] = *(const unsigned int*)&As[cur][m][a_koff(v, hf)];
        }
#pragma unroll
        for (int j = 0; j < 2; j++) {
            int n = wn0 + j * 16 + ln;
#pragma unroll
            for (int v = 0; v < 8; v++)
                bfr[j].u[v] = *(const unsigned int*)&Bt[cur][n][b_koff(v, hf)];
        }
#pragma unroll
        for (int i = 0; i < 4; i++)
#pragma unroll
            for (int j = 0; j < 2; j++)
                acc[i][j] = __builtin_amdgcn_wmma_f32_16x16x32_f16(
                    false, afr[i].h, false, bfr[j].h,
                    (short)0, acc[i][j], false, false);

        // ---- stage next tile into the other buffer ----
        if (hasNext) {
            int nxt = cur ^ 1;
            *(uint4*)&As[nxt][ar][ac]     = na0;
            *(uint4*)&As[nxt][ar][ac + 8] = na1;
#pragma unroll
            for (int e = 0; e < 16; e++) Bt[nxt][bcb + e][bkr] = nrb.h[e];
        }
        __syncthreads();
        cur ^= 1;
    }

    // ---- epilogue: C/D layout is row M = hf*8+r, col N = lane%16 ----
#pragma unroll
    for (int i = 0; i < 4; i++) {
#pragma unroll
        for (int j = 0; j < 2; j++) {
            const int col = bn0 + wn0 + j * 16 + ln;
            float bj = 0.0f;
            if (MODE != 0) bj = bias[col];
#pragma unroll
            for (int r = 0; r < 8; r++) {
                const int row = bm0 + wm0 + i * 16 + hf * 8 + r;
                float val = acc[i][j][r];
                if (MODE == 0) {
                    // col -> (three, head, d); row -> (b, n)
                    int three = col >> 10;
                    int head  = (col >> 6) & (HH - 1);
                    int dd    = col & (HD - 1);
                    int bidx  = row >> 11;           // / NN
                    int nrow  = row & (NN - 1);
                    size_t dst = (((size_t)(bidx * HH + head)) * NN + nrow) * HD + dd;
                    _Float16 hv = (_Float16)val;
                    if (three == 0)      qh[dst] = hv;
                    else if (three == 1) kh[dst] = hv;
                    else                 vh[dst] = hv;
                } else {
                    val += bj;
                    out_f32[(size_t)row * Nout + col] = val;
                    if (MODE == 1)
                        out_f16[(size_t)row * Nout + col] = (_Float16)val;
                }
            }
        }
    }
}

// ---------------------------------------------------------------------------
// Flash attention: one (b, h, 128-row q block) per workgroup; 8 wave32 waves,
// each wave owns one 16-row WMMA tile of Q. K/V processed in 64-key blocks
// with double-buffered LDS + software pipeline. Causal: mask-free main loop;
// the two diagonal-straddling blocks use a separately instantiated masked
// path (compile-time DIAG).
// ---------------------------------------------------------------------------
#define QB 128

__global__ __launch_bounds__(256) void attn_kernel(
    const _Float16* __restrict__ Q,   // [B,H,N,HD]
    const _Float16* __restrict__ Kg,  // [B,H,N,HD]
    const _Float16* __restrict__ Vg,  // [B,H,N,HD]
    _Float16* __restrict__ Oh)        // [B,N,C] f16
{
    __shared__ _Float16 Qs[QB][64];       // [q][d]
    __shared__ _Float16 Ks[2][64][64];    // [key][d]  (B-frag for S: pairs along d)
    __shared__ _Float16 Vt[2][64][64];    // [d][key]  (B-frag for O: pairs along key)
    __shared__ _Float16 Ps[8][16][64];    // per-wave P staging [q][key]

    const int tid  = threadIdx.x;
    const int lane = tid & 31;
    const int wave = tid >> 5;
    const int hf   = (lane >> 4) & 1;
    const int ln   = lane & 15;

    const int bid  = blockIdx.x;
    const int qblk = bid & 15;            // N/128 = 16 q blocks
    const int head = (bid >> 4) & (HH - 1);
    const int b    = bid >> 8;

    const size_t hoff = ((size_t)(b * HH + head)) * NN * HD;
    const _Float16* Qp = Q  + hoff;
    const _Float16* Kp = Kg + hoff;
    const _Float16* Vp = Vg + hoff;

    {   // load Q block: 128x64 halves; each thread 32 contiguous halves
        int r = tid >> 1;
        int c = (tid & 1) * 32;
        const uint4* s4 = (const uint4*)(Qp + (size_t)(qblk * QB + r) * HD + c);
        uint4* d4 = (uint4*)&Qs[r][c];
        d4[0] = s4[0]; d4[1] = s4[1]; d4[2] = s4[2]; d4[3] = s4[3];
    }

    // K/V staging coordinates: 64x64 halves, 256 threads -> 16 halves each
    const int kvr = tid >> 2;             // 0..63
    const int kvc = (tid & 3) * 16;       // 0,16,32,48
    union VU { uint4 u[2]; _Float16 h[16]; };

    __syncthreads();

    // Q A-fragments are loop-invariant: hoist them out of the k-block loop
    frag16 aq[2];
#pragma unroll
    for (int kk = 0; kk < 2; kk++) {
        int m = wave * 16 + ln;
#pragma unroll
        for (int v = 0; v < 8; v++)
            aq[kk].u[v] = *(const unsigned int*)&Qs[m][kk * 32 + a_koff(v, hf)];
    }

    v8f vzero = {};
    float m_r[8], l_r[8];
    v8f acc_o[4];
#pragma unroll
    for (int r = 0; r < 8; r++) { m_r[r] = -3.0e38f; l_r[r] = 0.f; }
#pragma unroll
    for (int d = 0; d < 4; d++) acc_o[d] = vzero;

    const int qrow_base = qblk * QB + wave * 16;
    const int nblk = 2 * qblk + 2;        // 64-key blocks covering [0, qblk*128+128)

    // ---- pipeline helpers ----
    auto load_kv = [&](int kb, uint4 kr[2], VU& vr) {
        const uint4* ks4 = (const uint4*)(Kp + (size_t)(kb * 64 + kvr) * HD + kvc);
        kr[0] = ks4[0]; kr[1] = ks4[1];
        const uint4* vs4 = (const uint4*)(Vp + (size_t)(kb * 64 + kvr) * HD + kvc);
        vr.u[0] = vs4[0]; vr.u[1] = vs4[1];
    };
    auto store_kv = [&](int buf, const uint4 kr[2], const VU& vr) {
        *(uint4*)&Ks[buf][kvr][kvc]     = kr[0];
        *(uint4*)&Ks[buf][kvr][kvc + 8] = kr[1];
#pragma unroll
        for (int e = 0; e < 16; e++) Vt[buf][kvc + e][kvr] = vr.h[e];
    };

    // ---- per-block compute: S = QK^T, online softmax, O += P V ----
    auto process = [&](int kblk, int buf, auto diagc) {
        constexpr bool DIAG = decltype(diagc)::value;

        v8f s[4];
#pragma unroll
        for (int j = 0; j < 4; j++) s[j] = vzero;
#pragma unroll
        for (int j = 0; j < 4; j++) {
#pragma unroll
            for (int kk = 0; kk < 2; kk++) {
                frag16 bb;
                int n = j * 16 + ln;
#pragma unroll
                for (int v = 0; v < 8; v++)
                    bb.u[v] = *(const unsigned int*)&Ks[buf][n][kk * 32 + b_koff(v, hf)];
                s[j] = __builtin_amdgcn_wmma_f32_16x16x32_f16(
                    false, aq[kk].h, false, bb.h, (short)0, s[j], false, false);
            }
        }

        // scale into log2 domain (+ causal mask on diagonal blocks only)
#pragma unroll
        for (int j = 0; j < 4; j++) {
            int key = kblk * 64 + j * 16 + ln;
#pragma unroll
            for (int r = 0; r < 8; r++) {
                float v = s[j][r] * SM_SCALE;
                if (DIAG) {
                    int qr = qrow_base + hf * 8 + r;
                    if (key > qr) v = -3.0e38f;
                }
                s[j][r] = v;
            }
        }

        // row max over 4 tiles + 16 lanes
        float alpha[8];
#pragma unroll
        for (int r = 0; r < 8; r++) {
            float v = fmaxf(fmaxf(s[0][r], s[1][r]), fmaxf(s[2][r], s[3][r]));
            v = fmaxf(v, __shfl_xor(v, 1, 32));
            v = fmaxf(v, __shfl_xor(v, 2, 32));
            v = fmaxf(v, __shfl_xor(v, 4, 32));
            v = fmaxf(v, __shfl_xor(v, 8, 32));
            float mn = fmaxf(m_r[r], v);
            alpha[r] = __builtin_amdgcn_exp2f(m_r[r] - mn);
            m_r[r]   = mn;
        }

        // P = exp2(S - m), stage to LDS (C-layout -> A-layout), row sums
        float rs[8];
#pragma unroll
        for (int r = 0; r < 8; r++) rs[r] = 0.f;
#pragma unroll
        for (int j = 0; j < 4; j++) {
#pragma unroll
            for (int r = 0; r < 8; r++) {
                float p = __builtin_amdgcn_exp2f(s[j][r] - m_r[r]);
                rs[r] += p;
                Ps[wave][hf * 8 + r][j * 16 + ln] = (_Float16)p;
            }
        }
#pragma unroll
        for (int r = 0; r < 8; r++) {
            float v = rs[r];
            v += __shfl_xor(v, 1, 32);
            v += __shfl_xor(v, 2, 32);
            v += __shfl_xor(v, 4, 32);
            v += __shfl_xor(v, 8, 32);
            l_r[r] = l_r[r] * alpha[r] + v;
        }
#pragma unroll
        for (int d = 0; d < 4; d++)
#pragma unroll
            for (int r = 0; r < 8; r++) acc_o[d][r] *= alpha[r];

        // Ps is same-wave write->read: DS ops execute in order within a wave
        // (no cross-wave sharing), so only a compiler fence is needed.
        asm volatile("" ::: "memory");

        // O += P @ V
#pragma unroll
        for (int d = 0; d < 4; d++) {
#pragma unroll
            for (int kk = 0; kk < 2; kk++) {
                frag16 a, bb;
#pragma unroll
                for (int v = 0; v < 8; v++)
                    a.u[v] = *(const unsigned int*)&Ps[wave][ln][kk * 32 + a_koff(v, hf)];
                int n = d * 16 + ln;
#pragma unroll
                for (int v = 0; v < 8; v++)
                    bb.u[v] = *(const unsigned int*)&Vt[buf][n][kk * 32 + b_koff(v, hf)];
                acc_o[d] = __builtin_amdgcn_wmma_f32_16x16x32_f16(
                    false, a.h, false, bb.h, (short)0, acc_o[d], false, false);
            }
        }
    };

    // ---- prologue: stage k-block 0 into buffer 0 ----
    {
        uint4 kr[2]; VU vr;
        load_kv(0, kr, vr);
        store_kv(0, kr, vr);
    }
    __syncthreads();

    int cur = 0;
    auto step = [&](int t, auto diagc) {
        const bool hasNext = (t + 1 < nblk);
        uint4 kr[2]; VU vr;
        if (hasNext) load_kv(t + 1, kr, vr);
        process(t, cur, diagc);
        if (hasNext) store_kv(cur ^ 1, kr, vr);
        __syncthreads();
        cur ^= 1;
    };

    int t = 0;
    for (; t < 2 * qblk; t++) step(t, BoolC<false>{});   // mask-free body
    for (; t < nblk; t++)     step(t, BoolC<true>{});    // diagonal blocks

    // ---- finalize: O / l, write f16 [b, n, head*HD + d] ----
#pragma unroll
    for (int r = 0; r < 8; r++) {
        float rl = 1.0f / l_r[r];
        int qr   = qrow_base + hf * 8 + r;
#pragma unroll
        for (int d = 0; d < 4; d++) {
            float v = acc_o[d][r] * rl;
            int col = head * HD + d * 16 + ln;
            Oh[((size_t)b * NN + qr) * CC + col] = (_Float16)v;
        }
    }
}

// ---------------------------------------------------------------------------
// Host launcher
// ---------------------------------------------------------------------------
extern "C" void kernel_launch(void* const* d_in, const int* in_sizes, int n_in,
                              void* d_out, int out_size, void* d_ws, size_t ws_size,
                              hipStream_t stream) {
    const float* x     = (const float*)d_in[0];
    // d_in[1] = mask (causality implemented structurally)
    const float* Wqkv  = (const float*)d_in[2];
    const float* Wproj = (const float*)d_in[3];
    const float* bproj = (const float*)d_in[4];
    const float* Wcaus = (const float*)d_in[5];
    const float* bcaus = (const float*)d_in[6];
    float* out = (float*)d_out;

    const int nX    = BB * NN * CC;      // 4194304
    const int nWqkv = CC * 3 * CC;       // 3145728
    const int nWp   = CC * CC;           // 1048576

    char* ws = (char*)d_ws;
    _Float16* x_h     = (_Float16*)(ws + 0);
    _Float16* wqkv_h  = (_Float16*)(ws + 8388608);
    _Float16* wproj_h = (_Float16*)(ws + 14680064);
    _Float16* wcaus_h = (_Float16*)(ws + 16777216);
    _Float16* q_h     = (_Float16*)(ws + 18874368);
    _Float16* k_h     = (_Float16*)(ws + 27262976);
    _Float16* v_h     = (_Float16*)(ws + 35651584);
    _Float16* attn_h  = (_Float16*)(ws + 44040192);
    _Float16* proj_h  = (_Float16*)(ws + 52428800);

    // 1) converts (vectorized x4)
    cvt_f16_kernel<<<(nX / 4 + 255) / 256, 256, 0, stream>>>(x, x_h, nX / 4);
    cvt_f16_kernel<<<(nWqkv / 4 + 255) / 256, 256, 0, stream>>>(Wqkv, wqkv_h, nWqkv / 4);
    cvt_f16_kernel<<<(nWp / 4 + 255) / 256, 256, 0, stream>>>(Wproj, wproj_h, nWp / 4);
    cvt_f16_kernel<<<(nWp / 4 + 255) / 256, 256, 0, stream>>>(Wcaus, wcaus_h, nWp / 4);

    // 2) qkv = x @ W_qkv, scattered to q/k/v [B,H,N,HD] f16
    {
        dim3 grid(3 * CC / BN, MROWS / BM);
        gemm_f16_kernel<0><<<grid, 256, 0, stream>>>(
            x_h, wqkv_h, CC, 3 * CC,
            nullptr, nullptr, nullptr, q_h, k_h, v_h);
    }

    // 3) causal flash attention -> attn_h [B,N,C] f16
    attn_kernel<<<BB * HH * (NN / QB), 256, 0, stream>>>(q_h, k_h, v_h, attn_h);

    // 4) out = attn @ W_proj + b_proj -> f32 d_out[0..] and f16 proj_h
    {
        dim3 grid(CC / BN, MROWS / BM);
        gemm_f16_kernel<1><<<grid, 256, 0, stream>>>(
            attn_h, wproj_h, CC, CC,
            bproj, out, proj_h, nullptr, nullptr, nullptr);
    }

    // 5) causal_features = out @ W_causal + b_causal -> f32 d_out[B*N*C..]
    {
        dim3 grid(CC / BN, MROWS / BM);
        gemm_f16_kernel<2><<<grid, 256, 0, stream>>>(
            proj_h, wcaus_h, CC, CC,
            bcaus, out + (size_t)MROWS * CC, nullptr, nullptr, nullptr, nullptr);
    }
}
